// ContinuumMemorySystem_5566277616153
// MI455X (gfx1250) — matrix operand order repaired and checked by reading, not compiled, not run on hardware
//
#include <hip/hip_runtime.h>
#include <math.h>

// ---------- problem constants ----------
#define DIMD   2048          // D
#define HIDH   8192          // H
#define BB     4
#define SS     2048
#define BST    (BB*SS)       // 8192 tokens
#define NPRED  8188          // B*(S-1)
#define NPAD   8192          // padded n (zero rows)
#define HD     ((size_t)HIDH*DIMD)   // 16,777,216

typedef __bf16 bf16;
typedef __attribute__((ext_vector_type(8)))  __bf16 v8bf;
typedef __attribute__((ext_vector_type(16))) __bf16 v16bf;
typedef __attribute__((ext_vector_type(8)))  float  v8f;

// ---------- CDNA5 async global->LDS copy, GVS mode (SGPR base + VGPR u32 offset) ----------
// vdst = LDS byte offset VGPR, voff = per-lane unsigned byte offset, saddr = uniform 64-bit base.
#define ASYNC_CP16_S(ldsoff, voff, saddr)                               \
  asm volatile("global_load_async_to_lds_b128 %0, %1, %2"               \
               :: "v"(ldsoff), "v"(voff), "s"(saddr) : "memory")
#define WAIT_ASYNC() asm volatile("s_wait_asynccnt 0x0" ::: "memory")

__device__ __forceinline__ unsigned lds_off(const void* p) {
  // LDS aperture: hardware uses addr[31:0] as the LDS offset.
  return (unsigned)(uintptr_t)p;
}

// ---------- WMMA fragment helpers (wave32, gfx1250 layouts) ----------
// A 16x32 bf16: lanes 0-15 hold row M=lane, K 0..15; lanes 16-31 K 16..31 (contiguous)
__device__ __forceinline__ v16bf frag_a(const bf16* p) {
  v8bf lo = *(const v8bf*)p;         // K ko..ko+7
  v8bf hi = *(const v8bf*)(p + 8);   // K ko+8..ko+15
  return __builtin_shufflevector(lo, hi, 0,1,2,3,4,5,6,7,8,9,10,11,12,13,14,15);
}
// B 32x16 bf16: lanes 0-15 hold col N=lane, K 0..15; lanes 16-31 K 16..31 (contiguous)
__device__ __forceinline__ v16bf frag_b(const bf16* p) {
  v8bf lo = *(const v8bf*)p;         // K kb..kb+7
  v8bf hi = *(const v8bf*)(p + 8);   // K kb+8..kb+15
  return __builtin_shufflevector(lo, hi, 0,1,2,3,4,5,6,7,8,9,10,11,12,13,14,15);
}
__device__ __forceinline__ v8f wmma_bf16(v16bf a, v16bf b, v8f c) {
  return __builtin_amdgcn_wmma_f32_16x16x32_bf16(false, a, false, b, (short)0, c, false, false);
}

// Tile config: block 128(M) x 64(N), BK=64 (two 32-K slices per buffer), 256 threads =
// 8 waves in 4(M) x 2(N); each wave computes 32x32 = 2x2 WMMA subtiles per slice.
// LDS rows padded to 40 halves; double-buffered; filled by async global->LDS DMA.
#define LDSP 40
#define ASL  (128 * LDSP * 2)   // A slice stride (bytes)
#define BSL  (64  * LDSP * 2)   // B slice stride (bytes)

// ---------- conversion kernels ----------
__global__ __launch_bounds__(256) void k_cvt_x(const float* __restrict__ x,
                                               bf16* __restrict__ xb,
                                               bf16* __restrict__ xt) {
  size_t gid = (size_t)blockIdx.x * 256 + threadIdx.x;      // over BST*D
  int t = (int)(gid >> 11);       // /DIMD
  int d = (int)(gid & (DIMD-1));
  float v = x[gid];
  xb[gid] = (bf16)v;
  int s = t & (SS-1);
  if (s < SS-1) {                  // pred token
    int i = t - (t >> 11);         // t - b
    xt[(size_t)d * NPAD + i] = (bf16)v;
  }
}

__global__ __launch_bounds__(256) void k_cvt_w(const float* __restrict__ Wg,
                                               const float* __restrict__ Wu,
                                               const float* __restrict__ Wd,
                                               bf16* __restrict__ Wg_b,
                                               bf16* __restrict__ Wu_b,
                                               bf16* __restrict__ Wd_b,
                                               bf16* __restrict__ Wdt_b) {
  size_t gid = (size_t)blockIdx.x * 256 + threadIdx.x;      // over H*D
  Wg_b[gid] = (bf16)Wg[gid];
  Wu_b[gid] = (bf16)Wu[gid];
  float wd = Wd[gid];              // Wd is [D,H]: gid = d*H + h
  Wd_b[gid] = (bf16)wd;
  size_t d = gid >> 13;            // /HIDH
  size_t h = gid & (HIDH-1);
  Wdt_b[h * DIMD + d] = (bf16)wd;  // W_down^T [H,D]
}

__global__ __launch_bounds__(256) void k_zero_tails(bf16* __restrict__ xt,
                                                    bf16* __restrict__ ht) {
  int gid = blockIdx.x * 256 + threadIdx.x;                  // (D+H)*4
  if (gid >= (DIMD + HIDH) * 4) return;
  int pad = gid & 3, row = gid >> 2;
  if (row < DIMD) xt[(size_t)row * NPAD + NPRED + pad] = (bf16)0.f;
  else            ht[(size_t)(row - DIMD) * NPAD + NPRED + pad] = (bf16)0.f;
}

// ---------- GEMM 1: fused gate+up (dual-B) with SwiGLU epilogue ----------
// A = xb [BST,D], Bg/Bu = W [H,D]; outputs bf16 gate_pre, up_pre, h; h^T at pred rows.
__global__ __launch_bounds__(256) void k_gemm_gateup(const bf16* __restrict__ A,
                                                     const bf16* __restrict__ Bg,
                                                     const bf16* __restrict__ Bu,
                                                     bf16* __restrict__ gp,
                                                     bf16* __restrict__ up,
                                                     bf16* __restrict__ hf,
                                                     bf16* __restrict__ ht) {
  __shared__ bf16 sA[2][2][128 * LDSP];
  __shared__ bf16 sG[2][2][64 * LDSP];
  __shared__ bf16 sU[2][2][64 * LDSP];
  const int tid = threadIdx.x, lane = tid & 31, wid = tid >> 5;
  const int wm = wid >> 1, wn = wid & 1;
  const int m0 = blockIdx.y * 128, n0 = blockIdx.x * 64;
  const int K = DIMD;
  v8f accG[2][2] = {}, accU[2][2] = {};
  const int ar = tid >> 1, ac = (tid & 1) * 16;
  const int br = tid >> 2, bc = (tid & 3) * 8;
  const int lr = lane & 15, ko = (lane >> 4) * 16, kb = (lane >> 4) * 16;

  const unsigned aO = (unsigned)((ar * LDSP + ac) * 2);
  const unsigned bO = (unsigned)((br * LDSP + bc) * 2);
  const unsigned ldsA[2] = { lds_off(&sA[0][0][0]), lds_off(&sA[1][0][0]) };
  const unsigned ldsG[2] = { lds_off(&sG[0][0][0]), lds_off(&sG[1][0][0]) };
  const unsigned ldsU[2] = { lds_off(&sU[0][0][0]), lds_off(&sU[1][0][0]) };
  // per-lane loop-invariant global byte offsets (GVS voffset)
  const unsigned vA0 = 2u * (unsigned)(ar * K + ac);
  const unsigned vB0 = 2u * (unsigned)(br * K + bc);
  // uniform 64-bit bases (advance by 2*k0 per slab)
  const unsigned long long baseA = (unsigned long long)(uintptr_t)A  + 2ull * (unsigned long long)m0 * K;
  const unsigned long long baseG = (unsigned long long)(uintptr_t)Bg + 2ull * (unsigned long long)n0 * K;
  const unsigned long long baseU = (unsigned long long)(uintptr_t)Bu + 2ull * (unsigned long long)n0 * K;

  auto fill = [&](int buf, int kk) {
    unsigned long long sa = baseA + 2ull * kk;
    unsigned long long sg = baseG + 2ull * kk;
    unsigned long long su = baseU + 2ull * kk;
    ASYNC_CP16_S(ldsA[buf] + aO,            vA0,       sa);
    ASYNC_CP16_S(ldsA[buf] + aO + 16,       vA0 + 16u, sa);
    ASYNC_CP16_S(ldsA[buf] + ASL + aO,      vA0 + 64u, sa);
    ASYNC_CP16_S(ldsA[buf] + ASL + aO + 16, vA0 + 80u, sa);
    ASYNC_CP16_S(ldsG[buf] + bO,            vB0,       sg);
    ASYNC_CP16_S(ldsG[buf] + BSL + bO,      vB0 + 64u, sg);
    ASYNC_CP16_S(ldsU[buf] + bO,            vB0,       su);
    ASYNC_CP16_S(ldsU[buf] + BSL + bO,      vB0 + 64u, su);
  };

  fill(0, 0);
  WAIT_ASYNC();
  __syncthreads();

  for (int k0 = 0; k0 < K; k0 += 64) {
    const int cur = (k0 >> 6) & 1, nxt = cur ^ 1;
    if (k0 + 64 < K) fill(nxt, k0 + 64);
#pragma unroll
    for (int sl = 0; sl < 2; ++sl) {
      v16bf a0 = frag_a(&sA[cur][sl][(wm*32 +      lr)*LDSP + ko]);
      v16bf a1 = frag_a(&sA[cur][sl][(wm*32 + 16 + lr)*LDSP + ko]);
      v16bf g0 = frag_b(&sG[cur][sl][(wn*32 +      lr)*LDSP + kb]);
      v16bf g1 = frag_b(&sG[cur][sl][(wn*32 + 16 + lr)*LDSP + kb]);
      v16bf u0 = frag_b(&sU[cur][sl][(wn*32 +      lr)*LDSP + kb]);
      v16bf u1 = frag_b(&sU[cur][sl][(wn*32 + 16 + lr)*LDSP + kb]);
      accG[0][0] = wmma_bf16(a0, g0, accG[0][0]);
      accG[0][1] = wmma_bf16(a0, g1, accG[0][1]);
      accG[1][0] = wmma_bf16(a1, g0, accG[1][0]);
      accG[1][1] = wmma_bf16(a1, g1, accG[1][1]);
      accU[0][0] = wmma_bf16(a0, u0, accU[0][0]);
      accU[0][1] = wmma_bf16(a0, u1, accU[0][1]);
      accU[1][0] = wmma_bf16(a1, u0, accU[1][0]);
      accU[1][1] = wmma_bf16(a1, u1, accU[1][1]);
    }
    WAIT_ASYNC();
    __syncthreads();
  }
  const int mrow = (lane >> 4) * 8;
  for (int mi = 0; mi < 2; ++mi)
    for (int ni = 0; ni < 2; ++ni)
      for (int v = 0; v < 8; ++v) {
        int t = m0 + wm*32 + mi*16 + mrow + v;
        int h = n0 + wn*32 + ni*16 + lr;
        float g = accG[mi][ni][v], u = accU[mi][ni][v];
        float sig = 1.f / (1.f + expf(-g));
        float hv = g * sig * u;
        size_t idx = (size_t)t * HIDH + h;
        gp[idx] = (bf16)g;  up[idx] = (bf16)u;  hf[idx] = (bf16)hv;
        int s = t & (SS-1);
        if (s < SS-1) { int i = t - (t >> 11); ht[(size_t)h * NPAD + i] = (bf16)hv; }
      }
}

// ---------- generic NT GEMM -> f32 (used for out + 3 grads) ----------
__global__ __launch_bounds__(256) void k_gemm_nt_f32(const bf16* __restrict__ A,
                                                     const bf16* __restrict__ Bm,
                                                     float* __restrict__ C,
                                                     int N, int K, float scale) {
  __shared__ bf16 sA[2][2][128 * LDSP];
  __shared__ bf16 sB[2][2][64 * LDSP];
  const int tid = threadIdx.x, lane = tid & 31, wid = tid >> 5;
  const int wm = wid >> 1, wn = wid & 1;
  const int m0 = blockIdx.y * 128, n0 = blockIdx.x * 64;
  v8f acc[2][2] = {};
  const int ar = tid >> 1, ac = (tid & 1) * 16;
  const int br = tid >> 2, bc = (tid & 3) * 8;
  const int lr = lane & 15, ko = (lane >> 4) * 16, kb = (lane >> 4) * 16;

  const unsigned aO = (unsigned)((ar * LDSP + ac) * 2);
  const unsigned bO = (unsigned)((br * LDSP + bc) * 2);
  const unsigned ldsA[2] = { lds_off(&sA[0][0][0]), lds_off(&sA[1][0][0]) };
  const unsigned ldsB[2] = { lds_off(&sB[0][0][0]), lds_off(&sB[1][0][0]) };
  const unsigned vA0 = 2u * (unsigned)(ar * K + ac);
  const unsigned vB0 = 2u * (unsigned)(br * K + bc);
  const unsigned long long baseA = (unsigned long long)(uintptr_t)A  + 2ull * (unsigned long long)m0 * K;
  const unsigned long long baseB = (unsigned long long)(uintptr_t)Bm + 2ull * (unsigned long long)n0 * K;

  auto fill = [&](int buf, int kk) {
    unsigned long long sa = baseA + 2ull * kk;
    unsigned long long sb = baseB + 2ull * kk;
    ASYNC_CP16_S(ldsA[buf] + aO,            vA0,       sa);
    ASYNC_CP16_S(ldsA[buf] + aO + 16,       vA0 + 16u, sa);
    ASYNC_CP16_S(ldsA[buf] + ASL + aO,      vA0 + 64u, sa);
    ASYNC_CP16_S(ldsA[buf] + ASL + aO + 16, vA0 + 80u, sa);
    ASYNC_CP16_S(ldsB[buf] + bO,            vB0,       sb);
    ASYNC_CP16_S(ldsB[buf] + BSL + bO,      vB0 + 64u, sb);
  };

  fill(0, 0);
  WAIT_ASYNC();
  __syncthreads();

  for (int k0 = 0; k0 < K; k0 += 64) {
    const int cur = (k0 >> 6) & 1, nxt = cur ^ 1;
    if (k0 + 64 < K) fill(nxt, k0 + 64);
#pragma unroll
    for (int sl = 0; sl < 2; ++sl) {
      v16bf a0 = frag_a(&sA[cur][sl][(wm*32 +      lr)*LDSP + ko]);
      v16bf a1 = frag_a(&sA[cur][sl][(wm*32 + 16 + lr)*LDSP + ko]);
      v16bf b0 = frag_b(&sB[cur][sl][(wn*32 +      lr)*LDSP + kb]);
      v16bf b1 = frag_b(&sB[cur][sl][(wn*32 + 16 + lr)*LDSP + kb]);
      acc[0][0] = wmma_bf16(a0, b0, acc[0][0]);
      acc[0][1] = wmma_bf16(a0, b1, acc[0][1]);
      acc[1][0] = wmma_bf16(a1, b0, acc[1][0]);
      acc[1][1] = wmma_bf16(a1, b1, acc[1][1]);
    }
    WAIT_ASYNC();
    __syncthreads();
  }
  const int mrow = (lane >> 4) * 8;
  for (int mi = 0; mi < 2; ++mi)
    for (int ni = 0; ni < 2; ++ni)
      for (int v = 0; v < 8; ++v) {
        int mm = m0 + wm*32 + mi*16 + mrow + v;
        int nn = n0 + wn*32 + ni*16 + lr;
        C[(size_t)mm * N + nn] = acc[mi][ni][v] * scale;
      }
}

// ---------- err = 2*(out[:, :-1] - x[:, 1:]); bf16 row-major + transposed; sumsq partials ----------
__global__ __launch_bounds__(256) void k_err(const float* __restrict__ outp,
                                             const float* __restrict__ x,
                                             bf16* __restrict__ err_b,
                                             bf16* __restrict__ errt,
                                             float* __restrict__ partials) {
  __shared__ float red[256];
  size_t gid = (size_t)blockIdx.x * 256 + threadIdx.x;  // over NPAD*D
  int i = (int)(gid >> 11), d = (int)(gid & (DIMD-1));
  float e = 0.f;
  if (i < NPRED) {
    int t = i + i / (SS-1);
    e = 2.f * (outp[(size_t)t * DIMD + d] - x[(size_t)(t+1) * DIMD + d]);
  }
  err_b[gid] = (bf16)e;
  errt[(size_t)d * NPAD + i] = (bf16)e;
  red[threadIdx.x] = e * e;
  __syncthreads();
  for (int off = 128; off > 0; off >>= 1) {
    if (threadIdx.x < off) red[threadIdx.x] += red[threadIdx.x + off];
    __syncthreads();
  }
  if (threadIdx.x == 0) partials[blockIdx.x] = red[0];
}

// ---------- d_h GEMM with SiLU-backward epilogue, transposed bf16 stores ----------
// A = err_b [NPAD,D], B = W_down^T [H,D]; epilogue -> dgp^T [H,NPAD], dup^T [H,NPAD]
__global__ __launch_bounds__(256) void k_gemm_dh(const bf16* __restrict__ A,
                                                 const bf16* __restrict__ Bm,
                                                 const bf16* __restrict__ gp,
                                                 const bf16* __restrict__ up,
                                                 bf16* __restrict__ dgpT,
                                                 bf16* __restrict__ dupT) {
  __shared__ bf16 sA[2][2][128 * LDSP];
  __shared__ bf16 sB[2][2][64 * LDSP];
  const int tid = threadIdx.x, lane = tid & 31, wid = tid >> 5;
  const int wm = wid >> 1, wn = wid & 1;
  const int m0 = blockIdx.y * 128, n0 = blockIdx.x * 64;
  const int K = DIMD;
  v8f acc[2][2] = {};
  const int ar = tid >> 1, ac = (tid & 1) * 16;
  const int br = tid >> 2, bc = (tid & 3) * 8;
  const int lr = lane & 15, ko = (lane >> 4) * 16, kb = (lane >> 4) * 16;

  const unsigned aO = (unsigned)((ar * LDSP + ac) * 2);
  const unsigned bO = (unsigned)((br * LDSP + bc) * 2);
  const unsigned ldsA[2] = { lds_off(&sA[0][0][0]), lds_off(&sA[1][0][0]) };
  const unsigned ldsB[2] = { lds_off(&sB[0][0][0]), lds_off(&sB[1][0][0]) };
  const unsigned vA0 = 2u * (unsigned)(ar * K + ac);
  const unsigned vB0 = 2u * (unsigned)(br * K + bc);
  const unsigned long long baseA = (unsigned long long)(uintptr_t)A  + 2ull * (unsigned long long)m0 * K;
  const unsigned long long baseB = (unsigned long long)(uintptr_t)Bm + 2ull * (unsigned long long)n0 * K;

  auto fill = [&](int buf, int kk) {
    unsigned long long sa = baseA + 2ull * kk;
    unsigned long long sb = baseB + 2ull * kk;
    ASYNC_CP16_S(ldsA[buf] + aO,            vA0,       sa);
    ASYNC_CP16_S(ldsA[buf] + aO + 16,       vA0 + 16u, sa);
    ASYNC_CP16_S(ldsA[buf] + ASL + aO,      vA0 + 64u, sa);
    ASYNC_CP16_S(ldsA[buf] + ASL + aO + 16, vA0 + 80u, sa);
    ASYNC_CP16_S(ldsB[buf] + bO,            vB0,       sb);
    ASYNC_CP16_S(ldsB[buf] + BSL + bO,      vB0 + 64u, sb);
  };

  fill(0, 0);
  WAIT_ASYNC();
  __syncthreads();

  for (int k0 = 0; k0 < K; k0 += 64) {
    const int cur = (k0 >> 6) & 1, nxt = cur ^ 1;
    if (k0 + 64 < K) fill(nxt, k0 + 64);
#pragma unroll
    for (int sl = 0; sl < 2; ++sl) {
      v16bf a0 = frag_a(&sA[cur][sl][(wm*32 +      lr)*LDSP + ko]);
      v16bf a1 = frag_a(&sA[cur][sl][(wm*32 + 16 + lr)*LDSP + ko]);
      v16bf b0 = frag_b(&sB[cur][sl][(wn*32 +      lr)*LDSP + kb]);
      v16bf b1 = frag_b(&sB[cur][sl][(wn*32 + 16 + lr)*LDSP + kb]);
      acc[0][0] = wmma_bf16(a0, b0, acc[0][0]);
      acc[0][1] = wmma_bf16(a0, b1, acc[0][1]);
      acc[1][0] = wmma_bf16(a1, b0, acc[1][0]);
      acc[1][1] = wmma_bf16(a1, b1, acc[1][1]);
    }
    WAIT_ASYNC();
    __syncthreads();
  }
  const int mrow = (lane >> 4) * 8;
  for (int mi = 0; mi < 2; ++mi)
    for (int ni = 0; ni < 2; ++ni)
      for (int v = 0; v < 8; ++v) {
        int i = m0 + wm*32 + mi*16 + mrow + v;
        int h = n0 + wn*32 + ni*16 + lr;
        float dgp = 0.f, duv = 0.f;
        if (i < NPRED) {
          int t = i + i / (SS-1);
          float dh = acc[mi][ni][v];
          float g = (float)gp[(size_t)t * HIDH + h];
          float u = (float)up[(size_t)t * HIDH + h];
          float sig = 1.f / (1.f + expf(-g));
          float act = g * sig;
          duv = dh * act;
          dgp = dh * u * sig * (1.f + g * (1.f - sig));
        }
        dgpT[(size_t)h * NPAD + i] = (bf16)dgp;
        dupT[(size_t)h * NPAD + i] = (bf16)duv;
      }
}

// ---------- deterministic reductions + scalar + final scaling ----------
__global__ __launch_bounds__(256) void k_sumsq(const float* __restrict__ src,
                                               size_t count, float* __restrict__ parts) {
  __shared__ float red[256];
  float s = 0.f;
  for (size_t idx = (size_t)blockIdx.x * 256 + threadIdx.x; idx < count;
       idx += (size_t)gridDim.x * 256) {
    float v = src[idx]; s += v * v;
  }
  red[threadIdx.x] = s;
  __syncthreads();
  for (int off = 128; off > 0; off >>= 1) {
    if (threadIdx.x < off) red[threadIdx.x] += red[threadIdx.x + off];
    __syncthreads();
  }
  if (threadIdx.x == 0) parts[blockIdx.x] = red[0];
}

__global__ __launch_bounds__(256) void k_final_sum(const float* __restrict__ parts,
                                                   int count, float* __restrict__ out) {
  __shared__ float red[256];
  float s = 0.f;
  for (int j = threadIdx.x; j < count; j += 256) s += parts[j];
  red[threadIdx.x] = s;
  __syncthreads();
  for (int off = 128; off > 0; off >>= 1) {
    if (threadIdx.x < off) red[threadIdx.x] += red[threadIdx.x + off];
    __syncthreads();
  }
  if (threadIdx.x == 0) *out = red[0];
}

__global__ void k_compute_sg(float* scal) {
  if (threadIdx.x == 0 && blockIdx.x == 0) {
    float em = sqrtf(scal[0]) / (float)NPRED;
    float ema = 0.95f * 1.0f + 0.05f * em;        // SURPRISE_EMA0 = 1.0
    scal[4] = fminf(em / fmaxf(ema, 1e-8f), 3.0f);
  }
}

__global__ __launch_bounds__(256) void k_scale_grads(float* __restrict__ grads,
                                                     const float* __restrict__ scal) {
  size_t gid = (size_t)blockIdx.x * 256 + threadIdx.x;  // over 3*HD
  int region = (int)(gid / HD);                          // 0=up,1=down,2=gate
  float sg = scal[4];
  float gn = sg * sqrtf(scal[1 + region]);
  float sc = sg * fminf(1.f, 1.f / fmaxf(gn, 1e-12f));
  grads[gid] *= sc;
}

// ---------- host launcher ----------
extern "C" void kernel_launch(void* const* d_in, const int* in_sizes, int n_in,
                              void* d_out, int out_size, void* d_ws, size_t ws_size,
                              hipStream_t stream) {
  const float* x  = (const float*)d_in[0];
  const float* Wu = (const float*)d_in[1];
  const float* Wg = (const float*)d_in[2];
  const float* Wd = (const float*)d_in[3];
  float* out_f   = (float*)d_out;                       // [BST, D]
  float* grad_up = out_f + (size_t)BST * DIMD;          // [H, D]
  float* grad_dn = grad_up + HD;                        // [D, H]
  float* grad_gt = grad_dn + HD;                        // [H, D]

  // workspace layout
  char* w = (char*)d_ws;
  auto alloc = [&](size_t bytes) { char* p = w; w += (bytes + 255) & ~(size_t)255; return p; };
  float* scal   = (float*)alloc(64 * 4);        // [0]=errss [1..3]=grad ss [4]=sg
  float* partsE = (float*)alloc(65536 * 4);
  float* partsG = (float*)alloc(3 * 4096 * 4);
  bf16* xb    = (bf16*)alloc((size_t)BST * DIMD * 2);
  bf16* xt    = (bf16*)alloc((size_t)DIMD * NPAD * 2);
  bf16* Wg_b  = (bf16*)alloc(HD * 2);
  bf16* Wu_b  = (bf16*)alloc(HD * 2);
  bf16* Wd_b  = (bf16*)alloc(HD * 2);
  bf16* Wdt_b = (bf16*)alloc(HD * 2);
  bf16* gp_b  = (bf16*)alloc((size_t)BST * HIDH * 2);
  bf16* up_b  = (bf16*)alloc((size_t)BST * HIDH * 2);
  bf16* hf_b  = (bf16*)alloc((size_t)BST * HIDH * 2);
  bf16* ht_b  = (bf16*)alloc((size_t)HIDH * NPAD * 2);
  bf16* err_b = (bf16*)alloc((size_t)NPAD * DIMD * 2);
  bf16* errt  = (bf16*)alloc((size_t)DIMD * NPAD * 2);
  bf16* dgpT  = (bf16*)alloc((size_t)HIDH * NPAD * 2);
  bf16* dupT  = (bf16*)alloc((size_t)HIDH * NPAD * 2);

  const float inv_n = 1.0f / (float)NPRED;

  // 0) conversions + tail zeroing
  k_cvt_x<<<(BST * (size_t)DIMD) / 256, 256, 0, stream>>>(x, xb, xt);
  k_cvt_w<<<HD / 256, 256, 0, stream>>>(Wg, Wu, Wd, Wg_b, Wu_b, Wd_b, Wdt_b);
  k_zero_tails<<<((DIMD + HIDH) * 4 + 255) / 256, 256, 0, stream>>>(xt, ht_b);

  // 1) gate/up dual GEMM + SwiGLU  (M=BST, N=H, K=D)
  k_gemm_gateup<<<dim3(HIDH / 64, BST / 128), 256, 0, stream>>>(
      xb, Wg_b, Wu_b, gp_b, up_b, hf_b, ht_b);

  // 2) forward out = h @ Wd^T  (M=BST, N=D, K=H) -> d_out
  k_gemm_nt_f32<<<dim3(DIMD / 64, BST / 128), 256, 0, stream>>>(
      hf_b, Wd_b, out_f, DIMD, HIDH, 1.0f);

  // 3) err (+ sumsq partials) -> err_b, errt
  k_err<<<((size_t)NPAD * DIMD) / 256, 256, 0, stream>>>(out_f, x, err_b, errt, partsE);
  k_final_sum<<<1, 256, 0, stream>>>(partsE, 65536, &scal[0]);
  k_compute_sg<<<1, 64, 0, stream>>>(scal);

  // 4) d_h GEMM + SiLU backward -> dgpT, dupT  (M=NPAD, N=H, K=D)
  k_gemm_dh<<<dim3(HIDH / 64, NPAD / 128), 256, 0, stream>>>(
      err_b, Wdt_b, gp_b, up_b, dgpT, dupT);

  // 5) grads (pre-clip, scaled by 1/n)
  // grad_down[D,H] = err^T @ h / n   (A=errt [D,NPAD], B=ht [H,NPAD])
  k_gemm_nt_f32<<<dim3(HIDH / 64, DIMD / 128), 256, 0, stream>>>(
      errt, ht_b, grad_dn, HIDH, NPAD, inv_n);
  // grad_gate[H,D] = dgp^T @ x / n   (A=dgpT [H,NPAD], B=xt [D,NPAD])
  k_gemm_nt_f32<<<dim3(DIMD / 64, HIDH / 128), 256, 0, stream>>>(
      dgpT, xt, grad_gt, DIMD, NPAD, inv_n);
  // grad_up[H,D] = dup^T @ x / n
  k_gemm_nt_f32<<<dim3(DIMD / 64, HIDH / 128), 256, 0, stream>>>(
      dupT, xt, grad_up, DIMD, NPAD, inv_n);

  // 6) grad norms (deterministic two-stage) + final clip scaling
  k_sumsq<<<4096, 256, 0, stream>>>(grad_up, HD, partsG + 0 * 4096);
  k_sumsq<<<4096, 256, 0, stream>>>(grad_dn, HD, partsG + 1 * 4096);
  k_sumsq<<<4096, 256, 0, stream>>>(grad_gt, HD, partsG + 2 * 4096);
  k_final_sum<<<1, 256, 0, stream>>>(partsG + 0 * 4096, 4096, &scal[1]);
  k_final_sum<<<1, 256, 0, stream>>>(partsG + 1 * 4096, 4096, &scal[2]);
  k_final_sum<<<1, 256, 0, stream>>>(partsG + 2 * 4096, 4096, &scal[3]);
  k_scale_grads<<<(3 * HD) / 256, 256, 0, stream>>>(grad_up, scal);
}
